// CodeArtSelfAttention_23794118819857
// MI455X (gfx1250) — matrix operand order repaired and checked by reading, compiled, not run on hardware
//
#include <hip/hip_runtime.h>

typedef _Float16 v16h __attribute__((ext_vector_type(16)));
typedef _Float16 v8h  __attribute__((ext_vector_type(8)));
typedef float    v8f  __attribute__((ext_vector_type(8)));
typedef unsigned int v4u __attribute__((ext_vector_type(4)));
typedef int      v8i  __attribute__((ext_vector_type(8)));
typedef int      v4i  __attribute__((ext_vector_type(4)));

#define BB 4
#define SS 2048
#define DM 768
#define HH 12
#define DH 64

// D = A(16x32 f16) * B(32x16 f16) + C(16x16 f32)
__device__ __forceinline__ v8f wmma16(v16h a, v16h b, v8f c) {
  return __builtin_amdgcn_wmma_f32_16x16x32_f16(false, a, false, b, (short)0, c,
                                                false, false);
}

// A-fragment: lane(half,l16) holds row l16,
// elements 0..15 -> K = half*16 + e
__device__ __forceinline__ v16h load_a(const _Float16* row, int half) {
  return *(const v16h*)(row + half * 16);
}

__device__ __forceinline__ unsigned lds_off(const void* p) {
  // generic pointer to LDS: low 32 bits are the LDS byte offset (aperture map)
  return (unsigned)(uintptr_t)p;
}

// TDM: load a 2D tile (tile0 x tile1 elements of 2 bytes) from global into LDS.
// D# packing per CDNA5 ISA 8.3/8.4: group0 = {count|flags, lds_addr, gaddr_lo,
// gaddr_hi|type=2}; group1 = {data_size, tensor dims, tile dims, dim0 stride}.
__device__ __forceinline__ void tdm_load_2d(unsigned lds, const _Float16* g,
                                            unsigned tile0, unsigned tile1,
                                            unsigned td0, unsigned td1,
                                            unsigned stride0_elems) {
  unsigned long long ga = (unsigned long long)(uintptr_t)g;
  v4u g0;
  g0[0] = 1u;                                   // count=1, user descriptor
  g0[1] = lds;                                  // lds_addr (bytes)
  g0[2] = (unsigned)ga;                         // global_addr[31:0]
  g0[3] = (unsigned)((ga >> 32) & 0x01ffffffu)  // global_addr[56:32]
          | 0x80000000u;                        // type=2 ("image") at [127:126]
  v8i g1;
  g1[0] = (int)(1u << 16);                      // wg_mask=0, data_size=1 (2B)
  g1[1] = (int)((td0 & 0xffffu) << 16);         // tensor_dim0[15:0]
  g1[2] = (int)(((td0 >> 16) & 0xffffu) | ((td1 & 0xffffu) << 16));
  g1[3] = (int)(((td1 >> 16) & 0xffffu) | ((tile0 & 0xffffu) << 16));
  g1[4] = (int)(tile1 & 0xffffu);               // tile_dim1 (tile_dim2=0)
  g1[5] = (int)stride0_elems;                   // tensor_dim0_stride[31:0]
  g1[6] = 0;                                    // stride0[47:32] | stride1[15:0]
  g1[7] = 0;
  v4i z4 = {0, 0, 0, 0};
  v8i z8 = {0, 0, 0, 0, 0, 0, 0, 0};
  // this toolchain declares the 6-arg form: (g0, g1, g2, g3, g_extra, cpol)
  __builtin_amdgcn_tensor_load_to_lds(g0, g1, z4, z4, z8, 0);
}

__global__ void __launch_bounds__(256) cvt_f32_f16(const float* __restrict__ src,
                                                   _Float16* __restrict__ dst, int n) {
  int i = blockIdx.x * blockDim.x + threadIdx.x;
  if (i < n) dst[i] = (_Float16)src[i];
}

// C[n,m] = sum_k X[n,k]*W[m,k] + bias[m]  (X:[8192,768] f16, W:[768,768] f16 row-major)
// vtrans==0 -> Out[b,h,s,dh] ; vtrans==1 -> Out[b,h,dh,s]
__global__ void __launch_bounds__(256) qkv_gemm(const _Float16* __restrict__ X,
                                                const _Float16* __restrict__ W,
                                                const float* __restrict__ bias,
                                                _Float16* __restrict__ Out,
                                                int vtrans) {
  const int lane = threadIdx.x & 31;
  const int wave = (blockIdx.x * blockDim.x + threadIdx.x) >> 5;
  const int mt = wave % (DM / 16);
  const int nt = wave / (DM / 16);
  const int half = lane >> 4, l16 = lane & 15;
  const int n0 = nt * 16, m0 = mt * 16;

  const _Float16* arow = X + (size_t)(n0 + l16) * DM;
  const _Float16* brow = W + (size_t)(m0 + l16) * DM;
  v8f acc = {};
#pragma unroll 4
  for (int k0 = 0; k0 < DM; k0 += 32) {
    v16h a = load_a(arow + k0, half);
    v16h b = *(const v16h*)(brow + k0 + half * 16);
    acc = wmma16(a, b, acc);
  }
  const int m = m0 + l16;
  const float bv = bias[m];
  const int h = m >> 6, dh = m & 63;
#pragma unroll
  for (int r = 0; r < 8; ++r) {
    const int n = n0 + half * 8 + r;
    const int bb = n >> 11, s = n & (SS - 1);
    size_t addr = vtrans ? (((size_t)(bb * HH + h) * DH + dh) * SS + s)
                         : (((size_t)(bb * HH + h) * SS + s) * DH + dh);
    Out[addr] = (_Float16)(acc[r] + bv);
  }
}

// Flash attention. Block = 8 waves sharing one (b,h); wave w owns q-tile
// (blockIdx%16)*8 + w. K/V 32-key chunks staged into LDS by the TDM
// (double-buffered), shared by all 8 waves.
__global__ void __launch_bounds__(256) attn_kernel(
    const _Float16* __restrict__ Qf,   // [B,H,S,Dh] f16
    const _Float16* __restrict__ Kf,   // [B,H,S,Dh] f16
    const _Float16* __restrict__ Vt,   // [B,H,Dh,S] f16
    const int* __restrict__ rpm,       // [B,S,S] int32
    const float* __restrict__ amask,   // [B,1,1,S]
    const float* __restrict__ relb,    // [9,H]
    float* __restrict__ out)           // [B,S,768]
{
  __shared__ __align__(128) _Float16 Kt[2][32 * DH];   // 32 keys x 64 dh
  __shared__ __align__(128) _Float16 Vb[2][DH * 32];   // 64 dh x 32 keys
  __shared__ __align__(64) _Float16 ldsP[8][16 * 32];
  __shared__ float ldsRow[8][16];

  const int lane = threadIdx.x & 31;
  const int wid = threadIdx.x >> 5;
  const int bh = blockIdx.x >> 4;          // 0..47
  const int h = bh % HH;
  const int b = bh / HH;
  const int q0 = ((blockIdx.x & 15) * 8 + wid) * 16;
  const int half = lane >> 4, l16 = lane & 15;

  const _Float16* qrow = Qf + ((size_t)(b * HH + h) * SS + q0 + l16) * DH;
  v16h qa0 = load_a(qrow + 0, half);
  v16h qa1 = load_a(qrow + 32, half);

  const _Float16* kbase = Kf + (size_t)(b * HH + h) * SS * DH;
  const _Float16* vbase = Vt + (size_t)(b * HH + h) * DH * SS;
  const int* rbase = rpm + (size_t)b * SS * SS;
  const float* mbase = amask + (size_t)b * SS;

  float rowM[8], rowL[8];
#pragma unroll
  for (int r = 0; r < 8; ++r) { rowM[r] = -1e30f; rowL[r] = 0.f; }
  v8f o0 = {}, o1 = {}, o2 = {}, o3 = {};

  _Float16* P = ldsP[wid];
  float* RW = ldsRow[wid];

  const int NCH = SS / 32;  // 64 chunks
  if (wid == 0) {
    // prologue: stage chunk 0 (K chunk is 4KB contiguous -> 1D tile)
    tdm_load_2d(lds_off(Kt[0]), kbase, 32 * DH, 1, SS * DH, 1, SS * DH);
    tdm_load_2d(lds_off(Vb[0]), vbase, 32, DH, SS, DH, SS);
  }

  for (int ch = 0; ch < NCH; ++ch) {
    const int k0 = ch * 32;
    const int cur = ch & 1;
    if (wid == 0) __builtin_amdgcn_s_wait_tensorcnt(0);  // chunk ch resident
    __syncthreads();  // publish chunk ch; fences prior reads of buf[cur^1]
    if (wid == 0 && ch + 1 < NCH) {
      const int k1 = k0 + 32;
      tdm_load_2d(lds_off(Kt[cur ^ 1]), kbase + (size_t)k1 * DH,
                  32 * DH, 1, SS * DH, 1, SS * DH);
      tdm_load_2d(lds_off(Vb[cur ^ 1]), vbase + k1, 32, DH, SS, DH, SS);
    }

    // ---- scores from LDS K tile: two 16x16 tiles, K-dim = Dh = 64
    const _Float16* Kl = Kt[cur];
    const _Float16* Vl = Vb[cur];
    v8f s0 = {}, s1 = {};
    {
      const _Float16* kr0 = Kl + (size_t)l16 * DH + half * 16;
      const _Float16* kr1 = Kl + (size_t)(16 + l16) * DH + half * 16;
      v16h bk;
      bk = *(const v16h*)(kr0);      s0 = wmma16(qa0, bk, s0);
      bk = *(const v16h*)(kr0 + 32); s0 = wmma16(qa1, bk, s0);
      bk = *(const v16h*)(kr1);      s1 = wmma16(qa0, bk, s1);
      bk = *(const v16h*)(kr1 + 32); s1 = wmma16(qa1, bk, s1);
    }

    // ---- scale + relative-position bias + additive mask
    const int key0 = k0 + l16, key1 = key0 + 16;
    const float mk0 = mbase[key0], mk1 = mbase[key1];
#pragma unroll
    for (int r = 0; r < 8; ++r) {
      const int q = q0 + half * 8 + r;
      const int* rr = rbase + (size_t)q * SS;
      int r0 = rr[key0]; r0 = (r0 >= 0) ? r0 : 8; r0 = (r0 < 8) ? r0 : 7;
      int r1 = rr[key1]; r1 = (r1 >= 0) ? r1 : 8; r1 = (r1 < 8) ? r1 : 7;
      s0[r] = s0[r] * 0.125f + relb[r0 * HH + h] + mk0;
      s1[r] = s1[r] * 0.125f + relb[r1 * HH + h] + mk1;
    }

    // ---- online softmax (row reductions across the 16 lanes of each half)
    float corr[8];
#pragma unroll
    for (int r = 0; r < 8; ++r) {
      float mx = fmaxf(s0[r], s1[r]);
      mx = fmaxf(mx, __shfl_xor(mx, 1, 32));
      mx = fmaxf(mx, __shfl_xor(mx, 2, 32));
      mx = fmaxf(mx, __shfl_xor(mx, 4, 32));
      mx = fmaxf(mx, __shfl_xor(mx, 8, 32));
      const float mnew = fmaxf(rowM[r], mx);
      corr[r] = __expf(rowM[r] - mnew);
      rowM[r] = mnew;
      const float p0 = __expf(s0[r] - mnew);
      const float p1 = __expf(s1[r] - mnew);
      s0[r] = p0; s1[r] = p1;
      float rs = p0 + p1;
      rs += __shfl_xor(rs, 1, 32);
      rs += __shfl_xor(rs, 2, 32);
      rs += __shfl_xor(rs, 4, 32);
      rs += __shfl_xor(rs, 8, 32);
      rowL[r] = rowL[r] * corr[r] + rs;
    }

    // ---- P (16q x 32k) -> LDS f16 (lane-layout transpose for the PV matmul)
#pragma unroll
    for (int r = 0; r < 8; ++r) {
      const int q = half * 8 + r;
      P[q * 32 + l16] = (_Float16)s0[r];
      P[q * 32 + 16 + l16] = (_Float16)s1[r];
    }
    if (l16 == 0) {
#pragma unroll
      for (int r = 0; r < 8; ++r) RW[half * 8 + r] = corr[r];
    }
    asm volatile("s_wait_dscnt 0" ::: "memory");  // wave-internal LDS RAW fence

    // ---- O^T = V^T x P^T : per-column (q) rescale is lane-uniform here
    const float cq = RW[l16];
    const v16h pb = *(const v16h*)(P + l16 * 32 + half * 16);
#pragma unroll
    for (int i = 0; i < 8; ++i) { o0[i] *= cq; o1[i] *= cq; o2[i] *= cq; o3[i] *= cq; }
    v16h va;
    va = load_a(Vl + (0 + l16) * 32, half);  o0 = wmma16(va, pb, o0);
    va = load_a(Vl + (16 + l16) * 32, half); o1 = wmma16(va, pb, o1);
    va = load_a(Vl + (32 + l16) * 32, half); o2 = wmma16(va, pb, o2);
    va = load_a(Vl + (48 + l16) * 32, half); o3 = wmma16(va, pb, o3);
  }

  // ---- normalize by row sums and store ctx[b, q, h*64 + dh]
  if (l16 == 0) {
#pragma unroll
    for (int r = 0; r < 8; ++r) RW[half * 8 + r] = rowL[r];
  }
  asm volatile("s_wait_dscnt 0" ::: "memory");
  const float linv = 1.0f / RW[l16];
  const int q = q0 + l16;
  float* orow = out + ((size_t)b * SS + q) * DM + h * DH;
#pragma unroll
  for (int r = 0; r < 8; ++r) {
    orow[0 + half * 8 + r]  = o0[r] * linv;
    orow[16 + half * 8 + r] = o1[r] * linv;
    orow[32 + half * 8 + r] = o2[r] * linv;
    orow[48 + half * 8 + r] = o3[r] * linv;
  }
}

extern "C" void kernel_launch(void* const* d_in, const int* in_sizes, int n_in,
                              void* d_out, int out_size, void* d_ws, size_t ws_size,
                              hipStream_t stream) {
  (void)in_sizes; (void)n_in; (void)out_size; (void)ws_size;
  const float* hs = (const float*)d_in[0];
  const float* am = (const float*)d_in[1];
  const int* rpm = (const int*)d_in[2];
  const float* Wq = (const float*)d_in[3];
  const float* bq = (const float*)d_in[4];
  const float* Wk = (const float*)d_in[5];
  const float* bk = (const float*)d_in[6];
  const float* Wv = (const float*)d_in[7];
  const float* bv = (const float*)d_in[8];
  const float* rb = (const float*)d_in[9];

  char* ws = (char*)d_ws;
  const size_t hsN = (size_t)BB * SS * DM;
  const size_t wN = (size_t)DM * DM;
  _Float16* hsF = (_Float16*)ws; ws += hsN * 2;
  _Float16* WqF = (_Float16*)ws; ws += wN * 2;
  _Float16* WkF = (_Float16*)ws; ws += wN * 2;
  _Float16* WvF = (_Float16*)ws; ws += wN * 2;
  _Float16* Qf  = (_Float16*)ws; ws += hsN * 2;
  _Float16* Kf  = (_Float16*)ws; ws += hsN * 2;
  _Float16* Vt  = (_Float16*)ws; ws += hsN * 2;

  cvt_f32_f16<<<(int)((hsN + 255) / 256), 256, 0, stream>>>(hs, hsF, (int)hsN);
  cvt_f32_f16<<<(int)((wN + 255) / 256), 256, 0, stream>>>(Wq, WqF, (int)wN);
  cvt_f32_f16<<<(int)((wN + 255) / 256), 256, 0, stream>>>(Wk, WkF, (int)wN);
  cvt_f32_f16<<<(int)((wN + 255) / 256), 256, 0, stream>>>(Wv, WvF, (int)wN);

  const int gemmBlocks = (512 * 48) / 8;
  qkv_gemm<<<gemmBlocks, 256, 0, stream>>>(hsF, WqF, bq, Qf, 0);
  qkv_gemm<<<gemmBlocks, 256, 0, stream>>>(hsF, WkF, bk, Kf, 0);
  qkv_gemm<<<gemmBlocks, 256, 0, stream>>>(hsF, WvF, bv, Vt, 1);

  const int attnBlocks = BB * HH * (SS / 128);  // 768: 8 q-tiles per block
  attn_kernel<<<attnBlocks, 256, 0, stream>>>(Qf, Kf, Vt, rpm, am, rb,
                                              (float*)d_out);
}
